// ProjectedGINRegressor_81552839016471
// MI455X (gfx1250) — compile-verified, matrix operand order checked by reading, not run on hardware
//
#include <hip/hip_runtime.h>
#include <hip/hip_bf16.h>

typedef __bf16 v16bf __attribute__((ext_vector_type(16)));
typedef __bf16 v8bf  __attribute__((ext_vector_type(8)));
typedef float  v8f   __attribute__((ext_vector_type(8)));

#define NNODES 100000
#define NEDGES 1600000
#define FDIM   128

// ---------- transpose 128x128 fp32 W[k][n] -> bf16 WT[n][k] ----------
__global__ void wt_bf16_kernel(const float* __restrict__ W, __bf16* __restrict__ WT) {
    int id = blockIdx.x * blockDim.x + threadIdx.x;   // 0..16383
    int k = id >> 7;
    int n = id & 127;
    WT[n * 128 + k] = (__bf16)W[k * 128 + n];
}

// ---------- agg = h ----------
__global__ void copy4_kernel(const float4* __restrict__ src, float4* __restrict__ dst, int n4) {
    int i = blockIdx.x * blockDim.x + threadIdx.x;
    if (i < n4) dst[i] = src[i];
}

// ---------- edge scatter: agg[dst] += h[src], one wave per edge ----------
__global__ void scatter_kernel(const float* __restrict__ h, const int* __restrict__ ei,
                               float* __restrict__ agg) {
    long long gid = (long long)blockIdx.x * blockDim.x + threadIdx.x;
    int e    = (int)(gid >> 5);
    int lane = (int)(gid & 31);
    if (e >= NEDGES) return;
    int src = ei[e];
    int dst = ei[NEDGES + e];
    const float4 v = *(const float4*)(h + (size_t)src * FDIM + lane * 4);
    float* a = agg + (size_t)dst * FDIM + lane * 4;
    unsafeAtomicAdd(a + 0, v.x);
    unsafeAtomicAdd(a + 1, v.y);
    unsafeAtomicAdd(a + 2, v.z);
    unsafeAtomicAdd(a + 3, v.w);
}

// ---------- fused 2-layer MLP with bf16 WMMA ----------
// out = relu( relu(z @ W1 + b1) @ W2 + b2 ), all 128x128, tile = 16 nodes/wave
__global__ __launch_bounds__(128)
void mlp_kernel(const float* __restrict__ z,
                const __bf16* __restrict__ w1T, const float* __restrict__ b1,
                const __bf16* __restrict__ w2T, const float* __restrict__ b2,
                float* __restrict__ out) {
    __shared__ __align__(16) __bf16 w_lds[128 * 128];     // 32 KB (W1 then W2)
    __shared__ __align__(16) __bf16 z1_lds[4][16 * 128];  // 16 KB, per-wave region

    const int tid       = threadIdx.x;
    const int wave      = tid >> 5;
    const int lane      = tid & 31;
    const int nlane     = lane & 15;
    const int half_sel  = (lane >> 4) & 1;   // lanes 16-31 take the +8 K half
    const int koff      = half_sel * 8;
    const int tile_base = (blockIdx.x * 4 + wave) * 16;

    // cooperative load W1^T into LDS (2048 x uint4)
    {
        const uint4* s = (const uint4*)w1T;
        uint4* d = (uint4*)w_lds;
#pragma unroll
        for (int i = 0; i < 16; ++i) d[tid + i * 128] = s[tid + i * 128];
    }
    __syncthreads();

    // A fragments for GEMM1 from global z (fp32 -> bf16), row M = nlane
    int rowm = tile_base + nlane;
    if (rowm >= NNODES) rowm = NNODES - 1;
    const float* zr = z + (size_t)rowm * FDIM;
    v16bf afrag[4];
#pragma unroll
    for (int kt = 0; kt < 4; ++kt) {
        const int K0 = kt * 32 + koff;
        float4 f0 = *(const float4*)(zr + K0);
        float4 f1 = *(const float4*)(zr + K0 + 4);
        float4 f2 = *(const float4*)(zr + K0 + 16);
        float4 f3 = *(const float4*)(zr + K0 + 20);
        v16bf a;
        a[0]=(__bf16)f0.x;  a[1]=(__bf16)f0.y;  a[2]=(__bf16)f0.z;  a[3]=(__bf16)f0.w;
        a[4]=(__bf16)f1.x;  a[5]=(__bf16)f1.y;  a[6]=(__bf16)f1.z;  a[7]=(__bf16)f1.w;
        a[8]=(__bf16)f2.x;  a[9]=(__bf16)f2.y;  a[10]=(__bf16)f2.z; a[11]=(__bf16)f2.w;
        a[12]=(__bf16)f3.x; a[13]=(__bf16)f3.y; a[14]=(__bf16)f3.z; a[15]=(__bf16)f3.w;
        afrag[kt] = a;
    }

    // GEMM1: z @ W1 + b1, relu -> z1_lds (bf16)
#pragma unroll
    for (int nt = 0; nt < 8; ++nt) {
        const float bv = b1[nt * 16 + nlane];
        v8f acc = {bv, bv, bv, bv, bv, bv, bv, bv};
#pragma unroll
        for (int kt = 0; kt < 4; ++kt) {
            const int K0   = kt * 32 + koff;
            const int ncol = nt * 16 + nlane;
            v8bf blo = *(const v8bf*)&w_lds[ncol * 128 + K0];
            v8bf bhi = *(const v8bf*)&w_lds[ncol * 128 + K0 + 16];
            v16bf bfrag = __builtin_shufflevector(blo, bhi,
                0,1,2,3,4,5,6,7,8,9,10,11,12,13,14,15);
            acc = __builtin_amdgcn_wmma_f32_16x16x32_bf16(false, afrag[kt], false, bfrag,
                                                          (short)0, acc, false, false);
        }
#pragma unroll
        for (int r = 0; r < 8; ++r) {
            const int mrow = r + 8 * half_sel;
            z1_lds[wave][mrow * 128 + nt * 16 + nlane] = (__bf16)fmaxf(acc[r], 0.0f);
        }
    }

    // swap in W2^T
    __syncthreads();
    {
        const uint4* s = (const uint4*)w2T;
        uint4* d = (uint4*)w_lds;
#pragma unroll
        for (int i = 0; i < 16; ++i) d[tid + i * 128] = s[tid + i * 128];
    }
    __syncthreads();

    // A fragments for GEMM2 from z1_lds
#pragma unroll
    for (int kt = 0; kt < 4; ++kt) {
        const int K0 = kt * 32 + koff;
        v8bf alo = *(const v8bf*)&z1_lds[wave][nlane * 128 + K0];
        v8bf ahi = *(const v8bf*)&z1_lds[wave][nlane * 128 + K0 + 16];
        afrag[kt] = __builtin_shufflevector(alo, ahi,
            0,1,2,3,4,5,6,7,8,9,10,11,12,13,14,15);
    }

    // GEMM2: z1 @ W2 + b2, relu -> out
#pragma unroll
    for (int nt = 0; nt < 8; ++nt) {
        const float bv = b2[nt * 16 + nlane];
        v8f acc = {bv, bv, bv, bv, bv, bv, bv, bv};
#pragma unroll
        for (int kt = 0; kt < 4; ++kt) {
            const int K0   = kt * 32 + koff;
            const int ncol = nt * 16 + nlane;
            v8bf blo = *(const v8bf*)&w_lds[ncol * 128 + K0];
            v8bf bhi = *(const v8bf*)&w_lds[ncol * 128 + K0 + 16];
            v16bf bfrag = __builtin_shufflevector(blo, bhi,
                0,1,2,3,4,5,6,7,8,9,10,11,12,13,14,15);
            acc = __builtin_amdgcn_wmma_f32_16x16x32_bf16(false, afrag[kt], false, bfrag,
                                                          (short)0, acc, false, false);
        }
#pragma unroll
        for (int r = 0; r < 8; ++r) {
            const int mrow = r + 8 * half_sel;
            const int row  = tile_base + mrow;
            if (row < NNODES)
                out[(size_t)row * FDIM + nt * 16 + nlane] = fmaxf(acc[r], 0.0f);
        }
    }
}

// ---------- final projection: out[i] = h[i] . w_out + b_out ----------
__global__ void proj_kernel(const float* __restrict__ h, const float* __restrict__ wout,
                            const float* __restrict__ bout, float* __restrict__ out) {
    long long gid = (long long)blockIdx.x * blockDim.x + threadIdx.x;
    int node = (int)(gid >> 5);
    int lane = (int)(gid & 31);
    if (node >= NNODES) return;
    float4 hv = *(const float4*)(h + (size_t)node * FDIM + lane * 4);
    float4 wv = *(const float4*)(wout + lane * 4);
    float s = hv.x * wv.x + hv.y * wv.y + hv.z * wv.z + hv.w * wv.w;
#pragma unroll
    for (int off = 16; off > 0; off >>= 1) s += __shfl_xor(s, off, 32);
    if (lane == 0) out[node] = s + bout[0];
}

extern "C" void kernel_launch(void* const* d_in, const int* in_sizes, int n_in,
                              void* d_out, int out_size, void* d_ws, size_t ws_size,
                              hipStream_t stream) {
    const float* x    = (const float*)d_in[0];
    const int*   ei   = (const int*)d_in[1];
    const float* w1[3] = {(const float*)d_in[2], (const float*)d_in[6], (const float*)d_in[10]};
    const float* B1[3] = {(const float*)d_in[3], (const float*)d_in[7], (const float*)d_in[11]};
    const float* w2[3] = {(const float*)d_in[4], (const float*)d_in[8], (const float*)d_in[12]};
    const float* B2[3] = {(const float*)d_in[5], (const float*)d_in[9], (const float*)d_in[13]};
    const float* wout = (const float*)d_in[14];
    const float* bout = (const float*)d_in[15];
    float* out = (float*)d_out;

    const size_t felems = (size_t)NNODES * FDIM;
    float*  agg = (float*)d_ws;
    float*  hA  = agg + felems;
    float*  hB  = hA + felems;
    __bf16* wt  = (__bf16*)(hB + felems);   // 6 matrices x 16384 bf16

    // convert + transpose all six MLP weight matrices to bf16
    for (int l = 0; l < 3; ++l) {
        wt_bf16_kernel<<<64, 256, 0, stream>>>(w1[l], wt + (size_t)(2 * l + 0) * 16384);
        wt_bf16_kernel<<<64, 256, 0, stream>>>(w2[l], wt + (size_t)(2 * l + 1) * 16384);
    }

    const int n4          = (int)(felems / 4);              // 3,200,000
    const int copy_blocks = (n4 + 255) / 256;
    const int scat_blocks = (NEDGES * 32 + 255) / 256;      // 200,000
    const int mlp_blocks  = (NNODES + 63) / 64;             // 1,563
    const int proj_blocks = (NNODES * 32 + 255) / 256;      // 12,500

    auto layer = [&](const float* hin, int l, float* hout) {
        copy4_kernel<<<copy_blocks, 256, 0, stream>>>((const float4*)hin, (float4*)agg, n4);
        scatter_kernel<<<scat_blocks, 256, 0, stream>>>(hin, ei, agg);
        mlp_kernel<<<mlp_blocks, 128, 0, stream>>>(agg,
                                                   wt + (size_t)(2 * l + 0) * 16384, B1[l],
                                                   wt + (size_t)(2 * l + 1) * 16384, B2[l],
                                                   hout);
    };

    layer(x,  0, hA);
    layer(hA, 1, hB);
    layer(hB, 2, hA);

    proj_kernel<<<proj_blocks, 256, 0, stream>>>(hA, wout, bout, out);
}
